// KEnhance_2499670966391
// MI455X (gfx1250) — matrix operand order repaired and checked by reading, not compiled, not run on hardware
//
#include <hip/hip_runtime.h>
#include <hip/hip_bf16.h>

typedef __attribute__((ext_vector_type(16))) _Float16 v16h;
typedef __attribute__((ext_vector_type(8)))  float    v8f;

#define N_ROWS 1024
#define T_STEPS 60
#define H_DIM 64
#define NEG_BIG  (-1.0e10f)
#define NEG_SMALL (-100000.0f)

__device__ __forceinline__ v8f wmma_f16(v16h a, v16h b, v8f c) {
  return __builtin_amdgcn_wmma_f32_16x16x32_f16(false, a, false, b, (short)0, c, false, false);
}
// fast transcendentals: v_rcp_f32 + v_exp_f32, no IEEE division chains
__device__ __forceinline__ float fast_rcp(float x) { return __builtin_amdgcn_rcpf(x); }
__device__ __forceinline__ float sigm(float x) { return fast_rcp(1.0f + __expf(-x)); }
__device__ __forceinline__ float fast_tanh(float x) {
  float e = __expf(-2.0f * x);
  return (1.0f - e) * fast_rcp(1.0f + e);
}
__device__ __forceinline__ void wait_async0() {
  asm volatile("s_wait_asynccnt 0x0" ::: "memory");
}

// ---------------------------------------------------------------------------
// prep: x [N,360] -> x_raw padded f16 [N,T,64] and h_in = tanh(x_raw@Wf^T+bf)
// ---------------------------------------------------------------------------
__global__ __launch_bounds__(256)
void prep_kernel(const float* __restrict__ x, const float* __restrict__ Wf,
                 const float* __restrict__ bf,
                 _Float16* __restrict__ xr16, _Float16* __restrict__ hin16) {
  long idx = (long)blockIdx.x * blockDim.x + threadIdx.x;     // over N*T*64
  if (idx >= (long)N_ROWS * T_STEPS * 64) return;
  int h = idx & 63;
  long nt = idx >> 6;
  int t = (int)(nt % T_STEPS);
  int n = (int)(nt / T_STEPS);
  float acc = bf[h];
#pragma unroll
  for (int d = 0; d < 6; ++d)
    acc += x[(long)n * 360 + d * 60 + t] * Wf[h * 6 + d];
  hin16[idx] = (_Float16)fast_tanh(acc);
  float xv = (h < 6) ? x[(long)n * 360 + h * 60 + t] : 0.0f;
  xr16[idx] = (_Float16)xv;
}

// ---------------------------------------------------------------------------
// GRU layer: one block owns 16 rows, runs the full T=60 recurrence.
// 384 threads = 12 waves = 12 N-tiles of the 192 gate outputs.
// B fragments (Wih/Whh) stay resident in VGPRs across all timesteps.
// Input tiles staged with double-buffered GLOBAL_LOAD_ASYNC_TO_LDS_B128,
// overlapped with the gate VALU work and published via s_wait_asynccnt.
// ---------------------------------------------------------------------------
__global__ __launch_bounds__(384)
void gru_layer_kernel(const _Float16* __restrict__ in_seq, long in_stride,
                      const float* __restrict__ Wih, long wih_stride,
                      const float* __restrict__ Whh, long whh_stride,
                      const float* __restrict__ bih, long bih_stride,
                      const float* __restrict__ bhh, long bhh_stride,
                      int wK,
                      _Float16* __restrict__ out_seq, long out_stride,
                      float* __restrict__ out_last, long last_stride) {
  const int inst = blockIdx.y;
  const int row0 = blockIdx.x * 16;
  const int tid = threadIdx.x;
  const int wave = tid >> 5;
  const int lane = tid & 31;

  __shared__ _Float16 xa[2][16][64];
  __shared__ _Float16 h16[16][64];
  __shared__ float    h32[16][64];
  __shared__ float    GI[16][192];
  __shared__ float    GH[16][192];
  __shared__ float    sbih[192];
  __shared__ float    sbhh[192];

  const _Float16* in_p = in_seq + in_stride * inst;
  const float* Wih_p = Wih + wih_stride * inst;
  const float* Whh_p = Whh + whh_stride * inst;

  for (int i = tid; i < 16 * 64; i += 384) {
    h16[i >> 6][i & 63] = (_Float16)0.0f;
    h32[i >> 6][i & 63] = 0.0f;
  }
  for (int i = tid; i < 192; i += 384) {
    sbih[i] = bih[bih_stride * inst + i];
    sbhh[i] = bhh[bhh_stride * inst + i];
  }

  // async x-tile loader: 128 lanes x 16B cover the 16x64 f16 tile
  auto issue_xtile = [&](int t, int buf) {
    if (tid < 128) {
      int r = tid >> 3, kc = (tid & 7) * 8;
      const _Float16* g = in_p + (((long)(row0 + r)) * T_STEPS + t) * 64 + kc;
      unsigned lds = (unsigned)(size_t)(&xa[buf][r][kc]);
      unsigned long long ga = (unsigned long long)(size_t)g;
      asm volatile("global_load_async_to_lds_b128 %0, %1, off"
                   :: "v"(lds), "v"(ga) : "memory");
    }
  };

  // Resident B fragments: B[k][n] = W[n_global][k] (W row-major [192][wK])
  v16h Bi0, Bi1, Bh0, Bh1;
  {
    const int col = wave * 16 + (lane & 15);
    const int kb = (lane & 16) ? 16 : 0;
#pragma unroll
    for (int e = 0; e < 16; ++e) {
      int k0 = kb + e, k1 = 32 + kb + e;
      Bi0[e] = (_Float16)((k0 < wK) ? Wih_p[col * wK + k0] : 0.0f);
      Bi1[e] = (_Float16)((k1 < wK) ? Wih_p[col * wK + k1] : 0.0f);
      Bh0[e] = (_Float16)Whh_p[col * 64 + k0];
      Bh1[e] = (_Float16)Whh_p[col * 64 + k1];
    }
  }
  issue_xtile(0, 0);
  wait_async0();
  __syncthreads();

  const int arow = lane & 15;
  const int m0 = (lane & 16) ? 8 : 0;
  const int ncol = wave * 16 + (lane & 15);

  for (int t = 0; t < T_STEPS; ++t) {
    const int cur = t & 1;
    v16h Ax0, Ax1, Ah0, Ah1;
#pragma unroll
    for (int e = 0; e < 16; ++e) {
      int j = e >> 1;
      int K = ((lane & 16) ? 16 : 0) + e;
      Ax0[e] = xa[cur][arow][K];
      Ax1[e] = xa[cur][arow][32 + K];
      Ah0[e] = h16[arow][K];
      Ah1[e] = h16[arow][32 + K];
    }
    v8f ci = {}; v8f ch = {};
    ci = wmma_f16(Ax0, Bi0, ci);
    ci = wmma_f16(Ax1, Bi1, ci);
    ch = wmma_f16(Ah0, Bh0, ch);
    ch = wmma_f16(Ah1, Bh1, ch);
#pragma unroll
    for (int r = 0; r < 8; ++r) {
      GI[m0 + r][ncol] = ci[r];
      GH[m0 + r][ncol] = ch[r];
    }
    if (t + 1 < T_STEPS) issue_xtile(t + 1, cur ^ 1);  // overlaps with gates
    __syncthreads();  // publish GI/GH; h16/xa[cur] reads complete

    for (int e = tid; e < 16 * 64; e += 384) {
      int m = e >> 6, j = e & 63;
      float gr = GI[m][j]        + sbih[j]        + GH[m][j]        + sbhh[j];
      float gz = GI[m][64 + j]   + sbih[64 + j]   + GH[m][64 + j]   + sbhh[64 + j];
      float rr = sigm(gr);
      float zz = sigm(gz);
      float gg = GI[m][128 + j] + sbih[128 + j] + rr * (GH[m][128 + j] + sbhh[128 + j]);
      float g = fast_tanh(gg);
      float hn = (1.0f - zz) * g + zz * h32[m][j];
      h32[m][j] = hn;
      h16[m][j] = (_Float16)hn;
      if (out_seq)
        out_seq[out_stride * inst + (((long)(row0 + m)) * T_STEPS + t) * 64 + j] = (_Float16)hn;
      if (out_last && t == T_STEPS - 1)
        out_last[last_stride * inst + ((long)(row0 + m)) * 64 + j] = hn;
    }
    wait_async0();    // issuing waves drain DMA before publishing xa[next]
    __syncthreads();  // publish h16 updates and next x-tile
  }
}

// ---------------------------------------------------------------------------
// finalize: normalize head features to f16, project x_last onto Wa/Wb (heads+rel)
// ---------------------------------------------------------------------------
__global__ __launch_bounds__(256)
void finalize_kernel(const float* __restrict__ fheads, const float* __restrict__ xlast,
                     const float* __restrict__ hWa, const float* __restrict__ hWb,
                     const float* __restrict__ rWa, const float* __restrict__ rWb,
                     _Float16* __restrict__ an16, float* __restrict__ aH,
                     float* __restrict__ bH, float* __restrict__ arel,
                     float* __restrict__ brel) {
  int idx = blockIdx.x * blockDim.x + threadIdx.x;  // 8*1024
  if (idx >= 8 * N_ROWS) return;
  int n = idx & (N_ROWS - 1);
  int s = idx >> 10;
  const float* xl = xlast + (long)n * 64;
  if (s < 7) {
    const float* f = fheads + ((long)s * N_ROWS + n) * 64;
    float nrm = 0.0f;
#pragma unroll
    for (int k = 0; k < 64; ++k) nrm += f[k] * f[k];
    float inv = fast_rcp(fmaxf(sqrtf(nrm), 1e-6f));
#pragma unroll
    for (int k = 0; k < 64; ++k)
      an16[((long)s * N_ROWS + n) * 64 + k] = (_Float16)(f[k] * inv);
    float a = 0.0f, b = 0.0f;
#pragma unroll
    for (int k = 0; k < 64; ++k) {
      a += xl[k] * hWa[s * 64 + k];
      b += xl[k] * hWb[s * 64 + k];
    }
    aH[s * N_ROWS + n] = a;
    bH[s * N_ROWS + n] = b;
  } else {
    float a = 0.0f, b = 0.0f;
#pragma unroll
    for (int k = 0; k < 64; ++k) {
      a += xl[k] * rWa[k];
      b += xl[k] * rWb[k];
    }
    arel[n] = a;
    brel[n] = b;
  }
}

// ---------------------------------------------------------------------------
// temporal attention + alstm: one block per ticker n
// ---------------------------------------------------------------------------
__global__ __launch_bounds__(64)
void att_kernel(const _Float16* __restrict__ xh, const float* __restrict__ W1,
                const float* __restrict__ b1, const float* __restrict__ W2,
                float* __restrict__ alstm) {
  __shared__ float sc[T_STEPS];
  __shared__ float xs[T_STEPS][64];
  int n = blockIdx.x, tid = threadIdx.x;
  for (int i = tid; i < T_STEPS * 64; i += 64)
    xs[i / 64][i % 64] = (float)xh[(long)n * T_STEPS * 64 + i];
  __syncthreads();
  if (tid < T_STEPS) {
    float s = 0.0f;
    for (int q = 0; q < 32; ++q) {
      float acc = b1[q];
#pragma unroll
      for (int k = 0; k < 64; ++k) acc += xs[tid][k] * W1[q * 64 + k];
      s += fast_tanh(acc) * W2[q];
    }
    sc[tid] = s;
  }
  __syncthreads();
  float mx = -1e30f;
  for (int t = 0; t < T_STEPS; ++t) mx = fmaxf(mx, sc[t]);
  float den = 0.0f;
  for (int t = 0; t < T_STEPS; ++t) den += __expf(sc[t] - mx);
  float acc = 0.0f;
  for (int t = 0; t < T_STEPS; ++t) acc += __expf(sc[t] - mx) * xs[t][tid];
  alstm[(long)n * 64 + tid] = acc * fast_rcp(den);
}

// ---------------------------------------------------------------------------
// head scores: S = an@an^T via WMMA (one wave per 16x16 tile), fuse leaky-relu
// gating and NEG_BIG masking. Sh: [7][N][N]
// ---------------------------------------------------------------------------
__global__ __launch_bounds__(32)
void head_score_kernel(const _Float16* __restrict__ an16, const float* __restrict__ aH,
                       const float* __restrict__ bH, const float* __restrict__ hb,
                       float* __restrict__ Sh) {
  int ti = blockIdx.x, tj = blockIdx.y, hd = blockIdx.z;
  int lane = threadIdx.x;
  const _Float16* an = an16 + (long)hd * N_ROWS * 64;

  v16h a0, a1, b0, b1;
  const int arow = ti * 16 + (lane & 15);
  const int bcol = tj * 16 + (lane & 15);
  const int kb = (lane & 16) ? 16 : 0;
#pragma unroll
  for (int e = 0; e < 16; ++e) {
    int j = e >> 1;
    int K = ((lane & 16) ? 16 : 0) + e;
    a0[e] = an[(long)arow * 64 + K];
    a1[e] = an[(long)arow * 64 + 32 + K];
    b0[e] = an[(long)bcol * 64 + kb + e];
    b1[e] = an[(long)bcol * 64 + 32 + kb + e];
  }
  v8f c = {};
  c = wmma_f16(a0, b0, c);
  c = wmma_f16(a1, b1, c);

  float bb = hb[hd];
  int m0 = (lane & 16) ? 8 : 0;
  int jn = tj * 16 + (lane & 15);
#pragma unroll
  for (int r = 0; r < 8; ++r) {
    int i = ti * 16 + r + m0;
    float g = c[r] - ((i == jn) ? 1.0f : 0.0f);
    float wv = aH[hd * N_ROWS + i] + bH[hd * N_ROWS + jn] + bb;
    wv = (wv > 0.0f) ? wv : 0.01f * wv;  // leaky_relu(0.01)
    float vw = (i == jn || g == 0.0f) ? NEG_BIG : g * wv;
    Sh[((long)hd * N_ROWS + i) * N_ROWS + jn] = vw;
  }
}

// ---------------------------------------------------------------------------
// relation pass: stream 256MB relation tensor once; fuse einsum + mask
// ---------------------------------------------------------------------------
__global__ __launch_bounds__(256)
void rel_score_kernel(const float* __restrict__ relm, const float* __restrict__ Wr,
                      const float* __restrict__ arel, const float* __restrict__ brel,
                      const float* __restrict__ bscal, float* __restrict__ tw,
                      unsigned char* __restrict__ mk) {
  long idx = (long)blockIdx.x * blockDim.x + threadIdx.x;  // N*N
  if (idx >= (long)N_ROWS * N_ROWS) return;
  int i = (int)(idx >> 10), j = (int)(idx & (N_ROWS - 1));
  const float4* rp = (const float4*)(relm + idx * 64);
  float dot = 0.0f, sm = 0.0f;
#pragma unroll
  for (int q = 0; q < 16; ++q) {
    float4 v = rp[q];
    const float* wr = Wr + q * 4;
    dot += v.x * wr[0] + v.y * wr[1] + v.z * wr[2] + v.w * wr[3];
    sm += v.x + v.y + v.z + v.w;
  }
  float wv = arel[i] + brel[j] + dot + bscal[0];
  wv = (wv > 0.0f) ? wv : 0.01f * wv;
  float m = (sm != 0.0f) ? 1.0f : 0.0f;
  float t = m * wv;
  tw[idx] = (t == 0.0f) ? NEG_SMALL : t;
  mk[idx] = (unsigned char)m;
}

// ---------------------------------------------------------------------------
// row softmax (+optional post-mask) then @ x_last : one block per row
// ---------------------------------------------------------------------------
__global__ __launch_bounds__(256)
void rowsm_kernel(const float* __restrict__ Sbase, long s_mat_stride,
                  const unsigned char* __restrict__ maskbase, long m_mat_stride,
                  const float* __restrict__ xlast, float* __restrict__ outbase,
                  long o_mat_stride) {
  int row = blockIdx.x, mat = blockIdx.y, tid = threadIdx.x;
  const float* S = Sbase + (long)mat * s_mat_stride + (long)row * N_ROWS;
  __shared__ float ev[N_ROWS];
  __shared__ float red[256];
  __shared__ float pacc[4][64];

  float mx = -1e30f;
  for (int j = tid; j < N_ROWS; j += 256) mx = fmaxf(mx, S[j]);
  red[tid] = mx;
  __syncthreads();
  for (int s = 128; s > 0; s >>= 1) {
    if (tid < s) red[tid] = fmaxf(red[tid], red[tid + s]);
    __syncthreads();
  }
  mx = red[0];
  __syncthreads();

  float ds = 0.0f;
  for (int j = tid; j < N_ROWS; j += 256) {
    float e = __expf(S[j] - mx);
    ev[j] = e;
    ds += e;
  }
  red[tid] = ds;
  __syncthreads();
  for (int s = 128; s > 0; s >>= 1) {
    if (tid < s) red[tid] += red[tid + s];
    __syncthreads();
  }
  float inv = fast_rcp(red[0]);
  __syncthreads();

  if (maskbase) {
    const unsigned char* M = maskbase + (long)mat * m_mat_stride + (long)row * N_ROWS;
    for (int j = tid; j < N_ROWS; j += 256) ev[j] = M[j] ? ev[j] : 0.0f;
    __syncthreads();
  }

  int h = tid & 63, c = tid >> 6;
  float acc = 0.0f;
  for (int j = c * 256; j < c * 256 + 256; ++j) acc += ev[j] * xlast[(long)j * 64 + h];
  pacc[c][h] = acc;
  __syncthreads();
  if (tid < 64) {
    float s = pacc[0][tid] + pacc[1][tid] + pacc[2][tid] + pacc[3][tid];
    outbase[(long)mat * o_mat_stride + (long)row * 64 + tid] = s * inv;
  }
}

// ---------------------------------------------------------------------------
// final FC: hidden = [x_last, alstm, head0..6, rel] (640) -> scalar per row
// ---------------------------------------------------------------------------
__global__ __launch_bounds__(64)
void fc_kernel(const float* __restrict__ xlast, const float* __restrict__ alstm,
               const float* __restrict__ headh, const float* __restrict__ relh,
               const float* __restrict__ Wo, const float* __restrict__ bo,
               float* __restrict__ out) {
  int n = blockIdx.x, tid = threadIdx.x;
  __shared__ float red[64];
  float acc = xlast[(long)n * 64 + tid] * Wo[tid];
  acc += alstm[(long)n * 64 + tid] * Wo[64 + tid];
#pragma unroll
  for (int h = 0; h < 7; ++h)
    acc += headh[((long)h * N_ROWS + n) * 64 + tid] * Wo[128 + h * 64 + tid];
  acc += relh[(long)n * 64 + tid] * Wo[576 + tid];
  red[tid] = acc;
  __syncthreads();
  for (int s = 32; s > 0; s >>= 1) {
    if (tid < s) red[tid] += red[tid + s];
    __syncthreads();
  }
  if (tid == 0) out[n] = red[0] + bo[0];
}

// ---------------------------------------------------------------------------
extern "C" void kernel_launch(void* const* d_in, const int* in_sizes, int n_in,
                              void* d_out, int out_size, void* d_ws, size_t ws_size,
                              hipStream_t stream) {
  // inputs: x, relation_matrix, then params pytree leaves (sorted dict keys)
  const float* x      = (const float*)d_in[0];
  const float* relm   = (const float*)d_in[1];
  const float* att_W1 = (const float*)d_in[2];   // att: (W1[32,64], b1[32], W2[1,32])
  const float* att_b1 = (const float*)d_in[3];
  const float* att_W2 = (const float*)d_in[4];
  const float* fc_Wo  = (const float*)d_in[5];   // fc: (Wo[1,640], bo[1])
  const float* fc_bo  = (const float*)d_in[6];
  const float* fin_Wf = (const float*)d_in[7];   // fc_in: (Wf[64,6], bf[64])
  const float* fin_bf = (const float*)d_in[8];
  const float* h_Wa   = (const float*)d_in[9];   // heads: Wa, Wb, b, gru0(4), gru1(4)
  const float* h_Wb   = (const float*)d_in[10];
  const float* h_b    = (const float*)d_in[11];
  const float* h0_Wih = (const float*)d_in[12];
  const float* h0_Whh = (const float*)d_in[13];
  const float* h0_bih = (const float*)d_in[14];
  const float* h0_bhh = (const float*)d_in[15];
  const float* h1_Wih = (const float*)d_in[16];
  const float* h1_Whh = (const float*)d_in[17];
  const float* h1_bih = (const float*)d_in[18];
  const float* h1_bhh = (const float*)d_in[19];
  const float* r_Wa   = (const float*)d_in[20];  // rel: (Wa, Wb, Wr, b)
  const float* r_Wb   = (const float*)d_in[21];
  const float* r_Wr   = (const float*)d_in[22];
  const float* r_b    = (const float*)d_in[23];
  const float* m0_Wih = (const float*)d_in[24];  // rnn: (l0(4), l1(4))
  const float* m0_Whh = (const float*)d_in[25];
  const float* m0_bih = (const float*)d_in[26];
  const float* m0_bhh = (const float*)d_in[27];
  const float* m1_Wih = (const float*)d_in[28];
  const float* m1_Whh = (const float*)d_in[29];
  const float* m1_bih = (const float*)d_in[30];
  const float* m1_bhh = (const float*)d_in[31];

  const long N = N_ROWS, T = T_STEPS;
  const long NT64 = N * T * 64;

  char* p = (char*)d_ws;
  auto alloc = [&](size_t b) -> void* {
    void* r = (void*)p;
    p += (b + 255) & ~(size_t)255;
    return r;
  };
  _Float16* xr16  = (_Float16*)alloc(NT64 * 2);
  _Float16* hin16 = (_Float16*)alloc(NT64 * 2);
  _Float16* seqA  = (_Float16*)alloc(8 * NT64 * 2);   // slot0=main, 1..7=heads
  _Float16* seqB  = (_Float16*)alloc(NT64 * 2);       // main layer-1 full sequence
  float* xlast    = (float*)alloc(N * 64 * 4);
  float* fheads   = (float*)alloc(7 * N * 64 * 4);
  _Float16* an16  = (_Float16*)alloc(7 * N * 64 * 2);
  float* aH       = (float*)alloc(7 * N * 4);
  float* bH       = (float*)alloc(7 * N * 4);
  float* arel     = (float*)alloc(N * 4);
  float* brel     = (float*)alloc(N * 4);
  float* alstm    = (float*)alloc(N * 64 * 4);
  float* Sh       = (float*)alloc(7L * N * N * 4);
  float* tw       = (float*)alloc((long)N * N * 4);
  unsigned char* mk = (unsigned char*)alloc((long)N * N);
  float* headh    = (float*)alloc(7 * N * 64 * 4);
  float* relh     = (float*)alloc(N * 64 * 4);
  (void)ws_size; (void)in_sizes; (void)n_in; (void)out_size;

  // 1) fc_in + input padding
  prep_kernel<<<(int)((N * T * 64 + 255) / 256), 256, 0, stream>>>(
      x, fin_Wf, fin_bf, xr16, hin16);

  // 2) GRU layer 0 — main (din=64) and 7 heads (din=6, K-padded)
  gru_layer_kernel<<<dim3(64, 1), 384, 0, stream>>>(
      hin16, 0, m0_Wih, 0, m0_Whh, 0, m0_bih, 0, m0_bhh, 0, 64,
      seqA, 0, (float*)nullptr, 0);
  gru_layer_kernel<<<dim3(64, 7), 384, 0, stream>>>(
      xr16, 0, h0_Wih, 192 * 6, h0_Whh, 192 * 64, h0_bih, 192, h0_bhh, 192, 6,
      seqA + NT64, NT64, (float*)nullptr, 0);

  // 3) GRU layer 1 — main emits full sequence + x_last; heads emit final h only
  gru_layer_kernel<<<dim3(64, 1), 384, 0, stream>>>(
      seqA, 0, m1_Wih, 0, m1_Whh, 0, m1_bih, 0, m1_bhh, 0, 64,
      seqB, 0, xlast, 0);
  gru_layer_kernel<<<dim3(64, 7), 384, 0, stream>>>(
      seqA + NT64, NT64, h1_Wih, 192 * 64, h1_Whh, 192 * 64, h1_bih, 192,
      h1_bhh, 192, 64, (_Float16*)nullptr, 0, fheads, N * 64);

  // 4) normalize head features, project x_last onto attention vectors
  finalize_kernel<<<(8 * 1024 + 255) / 256, 256, 0, stream>>>(
      fheads, xlast, h_Wa, h_Wb, r_Wa, r_Wb, an16, aH, bH, arel, brel);

  // 5) temporal attention -> alstm
  att_kernel<<<1024, 64, 0, stream>>>(seqB, att_W1, att_b1, att_W2, alstm);

  // 6) head similarity scores (WMMA) fused with gating/masking
  head_score_kernel<<<dim3(64, 64, 7), 32, 0, stream>>>(an16, aH, bH, h_b, Sh);

  // 7) relation tensor streaming pass (256 MB)
  rel_score_kernel<<<(int)((N * N + 255) / 256), 256, 0, stream>>>(
      relm, r_Wr, arel, brel, r_b, tw, mk);

  // 8) row softmax @ x_last (7 heads, then masked rel)
  rowsm_kernel<<<dim3(1024, 7), 256, 0, stream>>>(
      Sh, (long)N * N, (const unsigned char*)nullptr, 0, xlast, headh, N * 64);
  rowsm_kernel<<<dim3(1024, 1), 256, 0, stream>>>(
      tw, 0, mk, 0, xlast, relh, 0);

  // 9) final FC
  fc_kernel<<<1024, 64, 0, stream>>>(xlast, alstm, headh, relh, fc_Wo, fc_bo,
                                     (float*)d_out);
}
